// PWCNet_40132174414276
// MI455X (gfx1250) — matrix operand order repaired and checked by reading, not compiled, not run on hardware
//
#include <hip/hip_runtime.h>

// ---------------------------------------------------------------------------
// PWC-Net forward for MI455X (gfx1250, wave32).
// Convolutions via implicit-GEMM WMMA (v_wmma_f32_16x16x32_bf16):
//  - weights pre-packed to bf16 [Cout_pad64][K_pad32] (zero-filled)
//  - A fragments: two 16B global loads per 16x16 tile, no bounds checks
//  - B fragments: branchless clamped im2col gather, built in registers
//  - each wave: 64 output channels x 32 pixels -> 8 v_wmma per 32-K chunk
// ---------------------------------------------------------------------------

#define DIV_FLOW 0.05f

typedef __bf16 bf16_t;
typedef bf16_t v16bf __attribute__((ext_vector_type(16)));
typedef bf16_t v8bf  __attribute__((ext_vector_type(8)));
typedef float  v8f   __attribute__((ext_vector_type(8)));

#define CONV_WAVES 4   // waves per block; each wave owns a 32-pixel tile pair

static __device__ __forceinline__ v16bf load_a_frag(const bf16_t* __restrict__ wp,
                                                    long row, int Kp, int k0, int ah) {
  // A 16x32 bf16 layout: lane m holds K = {ah..ah+15}
  const v8bf* p = (const v8bf*)(wp + row * Kp + k0 + ah);
  v8bf lo = p[0];
  v8bf hi = p[1];            // +8 bf16
  v16bf r;
  #pragma unroll
  for (int e = 0; e < 8; ++e) { r[e] = lo[e]; r[e + 8] = hi[e]; }
  return r;
}

template <int KSZ>
__global__ __launch_bounds__(32 * CONV_WAVES) void conv2d_wmma_kernel(
    const float* __restrict__ in, const bf16_t* __restrict__ wp,
    const float* __restrict__ bias, float* __restrict__ out,
    int B, int Cin, int H, int W, int Cout, int Hout, int Wout,
    int stride, int dil, int pad, int relu, int coff, int Ctot, int Kp)
{
  const int lane = threadIdx.x & 31;
  const int wave = threadIdx.x >> 5;
  const long tileN = (long)blockIdx.x * CONV_WAVES + wave;
  const int mbase = blockIdx.y * 64;             // 4 tiles of 16 rows
  const long Npix = (long)B * Hout * Wout;

  // ---- per-lane B columns: two pixel tiles (n and n+16) ----
  const int bn = lane & 15;
  const long pix0 = tileN * 32 + bn;
  const long pix1 = pix0 + 16;
  const bool pv0 = pix0 < Npix;
  const bool pv1 = pix1 < Npix;
  long pc0 = pv0 ? pix0 : 0;
  long pc1 = pv1 ? pix1 : 0;
  const int ox0 = (int)(pc0 % Wout); pc0 /= Wout;
  const int oy0 = (int)(pc0 % Hout);
  const int bb0 = (int)(pc0 / Hout);
  const int ox1 = (int)(pc1 % Wout); pc1 /= Wout;
  const int oy1 = (int)(pc1 % Hout);
  const int bb1 = (int)(pc1 / Hout);
  const int iyb0 = oy0 * stride - pad, ixb0 = ox0 * stride - pad;
  const int iyb1 = oy1 * stride - pad, ixb1 = ox1 * stride - pad;
  const float* __restrict__ ib0 = in + (long)bb0 * Cin * H * W;
  const float* __restrict__ ib1 = in + (long)bb1 * Cin * H * W;

  const int kb = (lane >> 4) * 16;               // lanes 16-31: K+16
  const int arow = lane & 15;
  const int ah = (lane >> 4) * 16;               // lanes 16-31: K+16

  v8f acc00 = {}, acc10 = {}, acc20 = {}, acc30 = {};
  v8f acc01 = {}, acc11 = {}, acc21 = {}, acc31 = {};

  for (int k0 = 0; k0 < Kp; k0 += 32) {
    // ---- A fragments (bf16-packed, padded: unconditional 16B loads) ----
    v16bf a0 = load_a_frag(wp, (long)(mbase + 0  + arow), Kp, k0, ah);
    v16bf a1 = load_a_frag(wp, (long)(mbase + 16 + arow), Kp, k0, ah);
    v16bf a2 = load_a_frag(wp, (long)(mbase + 32 + arow), Kp, k0, ah);
    v16bf a3 = load_a_frag(wp, (long)(mbase + 48 + arow), Kp, k0, ah);

    // ---- B fragments: 16 consecutive K per lane, branchless gather ----
    int ks = k0 + kb;
    int ci = ks / (KSZ * KSZ);
    int rem = ks - ci * (KSZ * KSZ);
    int ky = rem / KSZ;
    int kx = rem - ky * KSZ;
    v16bf b0, b1;
    #pragma unroll
    for (int e = 0; e < 16; ++e) {
      const int dy = ky * dil, dx = kx * dil;
      const int cic = (ci < Cin) ? ci : (Cin - 1);
      const bool cok = (ci < Cin);
      const long cbase = (long)cic * H;

      int iy = iyb0 + dy, ix = ixb0 + dx;
      bool ok = pv0 && cok && (iy >= 0) && (iy < H) && (ix >= 0) && (ix < W);
      int iyc = min(max(iy, 0), H - 1);
      int ixc = min(max(ix, 0), W - 1);
      float v = ib0[(cbase + iyc) * W + ixc];         // always in-bounds
      b0[e] = (bf16_t)(ok ? v : 0.f);                 // v_cndmask, no branch

      iy = iyb1 + dy; ix = ixb1 + dx;
      ok = pv1 && cok && (iy >= 0) && (iy < H) && (ix >= 0) && (ix < W);
      iyc = min(max(iy, 0), H - 1);
      ixc = min(max(ix, 0), W - 1);
      v = ib1[(cbase + iyc) * W + ixc];
      b1[e] = (bf16_t)(ok ? v : 0.f);

      ++kx;
      if (kx == KSZ) { kx = 0; ++ky; if (ky == KSZ) { ky = 0; ++ci; } }
    }

    acc00 = __builtin_amdgcn_wmma_f32_16x16x32_bf16(false, a0, false, b0,
                                                    (short)0, acc00, false, false);
    acc10 = __builtin_amdgcn_wmma_f32_16x16x32_bf16(false, a1, false, b0,
                                                    (short)0, acc10, false, false);
    acc20 = __builtin_amdgcn_wmma_f32_16x16x32_bf16(false, a2, false, b0,
                                                    (short)0, acc20, false, false);
    acc30 = __builtin_amdgcn_wmma_f32_16x16x32_bf16(false, a3, false, b0,
                                                    (short)0, acc30, false, false);
    acc01 = __builtin_amdgcn_wmma_f32_16x16x32_bf16(false, a0, false, b1,
                                                    (short)0, acc01, false, false);
    acc11 = __builtin_amdgcn_wmma_f32_16x16x32_bf16(false, a1, false, b1,
                                                    (short)0, acc11, false, false);
    acc21 = __builtin_amdgcn_wmma_f32_16x16x32_bf16(false, a2, false, b1,
                                                    (short)0, acc21, false, false);
    acc31 = __builtin_amdgcn_wmma_f32_16x16x32_bf16(false, a3, false, b1,
                                                    (short)0, acc31, false, false);
  }

  // ---- epilogue: C/D layout — VGPR r: M=r (lanes 0-15) / M=r+8 (16-31) ----
  const int mh = (lane >> 4) * 8;
  const long hw = (long)Hout * Wout;
  #pragma unroll
  for (int np = 0; np < 2; ++np) {
    const bool pv = np ? pv1 : pv0;
    if (!pv) continue;
    const int bb = np ? bb1 : bb0;
    const int oy = np ? oy1 : oy0;
    const int ox = np ? ox1 : ox0;
    long obase = ((long)bb * Ctot + coff) * hw + (long)oy * Wout + ox;
    #pragma unroll
    for (int t = 0; t < 4; ++t) {
      v8f acc;
      if (np == 0) acc = (t == 0) ? acc00 : (t == 1) ? acc10 : (t == 2) ? acc20 : acc30;
      else         acc = (t == 0) ? acc01 : (t == 1) ? acc11 : (t == 2) ? acc21 : acc31;
      #pragma unroll
      for (int r = 0; r < 8; ++r) {
        int oc = mbase + t * 16 + r + mh;
        if (oc < Cout) {
          float v = acc[r] + bias[oc];
          if (relu) v = (v > 0.f) ? v : 0.1f * v;
          out[obase + (long)oc * hw] = v;
        }
      }
    }
  }
}

// Repack f32 OIHW weights -> bf16 [Coutp][Kp], zero-padded.
__global__ void pack_weights_kernel(const float* __restrict__ w,
                                    bf16_t* __restrict__ out,
                                    int Cout, int K, int Coutp, int Kp)
{
  long idx = (long)blockIdx.x * blockDim.x + threadIdx.x;
  long total = (long)Coutp * Kp;
  if (idx >= total) return;
  int k = (int)(idx % Kp);
  int oc = (int)(idx / Kp);
  float v = (oc < Cout && k < K) ? w[(long)oc * K + k] : 0.f;
  out[idx] = (bf16_t)v;
}

// ---------------------------------------------------------------------------
// Helper kernels
// ---------------------------------------------------------------------------

__global__ void warp_kernel(const float* __restrict__ x2,
                            const float* __restrict__ flow,
                            float* __restrict__ out,
                            int B, int C, int H, int W, int h_im, int w_im)
{
  long idx = (long)blockIdx.x * blockDim.x + threadIdx.x;
  long total = (long)B * C * H * W;
  if (idx >= total) return;
  long p = idx;
  int x = (int)(p % W); p /= W;
  int y = (int)(p % H); p /= H;
  int c = (int)(p % C); int b = (int)(p / C);
  long hw = (long)H * W;
  float sx = 2.0f / ((float)(w_im - 1) * DIV_FLOW);
  float sy = 2.0f / ((float)(h_im - 1) * DIV_FLOW);
  float fx = flow[((long)b * 2 + 0) * hw + (long)y * W + x] * sx;
  float fy = flow[((long)b * 2 + 1) * hw + (long)y * W + x] * sy;
  float gx = -1.f + 2.f * (float)x / (float)(W - 1);
  float gy = -1.f + 2.f * (float)y / (float)(H - 1);
  float px = (gx + fx + 1.f) * 0.5f * (float)(W - 1);
  float py = (gy + fy + 1.f) * 0.5f * (float)(H - 1);
  float x0f = floorf(px), y0f = floorf(py);
  int x0 = (int)x0f, y0 = (int)y0f;
  float wx1 = px - x0f, wx0 = 1.f - wx1;
  float wy1 = py - y0f, wy0 = 1.f - wy1;
  const float* img = x2 + ((long)b * C + c) * hw;
  float s = 0.f, ms = 0.f;
  #pragma unroll
  for (int dy = 0; dy < 2; ++dy) {
    #pragma unroll
    for (int dx = 0; dx < 2; ++dx) {
      int xi = x0 + dx, yi = y0 + dy;
      float wgt = (dx ? wx1 : wx0) * (dy ? wy1 : wy0);
      bool ok = (xi >= 0) && (xi < W) && (yi >= 0) && (yi < H);
      int xc = min(max(xi, 0), W - 1);
      int yc = min(max(yi, 0), H - 1);
      float v = img[(long)yc * W + xc];
      if (ok) { s += v * wgt; ms += wgt; }
    }
  }
  float mask = (ms >= 1.0f) ? 1.f : 0.f;
  out[idx] = s * mask;
}

// out channels [0,81) of a Ctot-channel tensor; leaky-ReLU fused.
__global__ void cost_volume_kernel(const float* __restrict__ f1,
                                   const float* __restrict__ f2,
                                   float* __restrict__ out,
                                   int B, int C, int H, int W, int Ctot)
{
  long idx = (long)blockIdx.x * blockDim.x + threadIdx.x;
  long total = (long)B * 81 * H * W;
  if (idx >= total) return;
  long p = idx;
  int x = (int)(p % W); p /= W;
  int y = (int)(p % H); p /= H;
  int d = (int)(p % 81); int b = (int)(p / 81);
  int i = d / 9, j = d % 9;
  int y2 = y + i - 4, x2 = x + j - 4;
  float s = 0.f;
  if (y2 >= 0 && y2 < H && x2 >= 0 && x2 < W) {
    long hw = (long)H * W;
    const float* a  = f1 + ((long)b * C) * hw + (long)y * W + x;
    const float* bb = f2 + ((long)b * C) * hw + (long)y2 * W + x2;
    for (int c = 0; c < C; ++c) s += a[(long)c * hw] * bb[(long)c * hw];
    s /= (float)C;
  }
  s = (s > 0.f) ? s : 0.1f * s;
  out[(((long)b * Ctot + d) * H + y) * W + x] = s;
}

__global__ void upsample_kernel(const float* __restrict__ src,
                                float* __restrict__ dst,
                                int B, int C, int Hs, int Ws, int Hd, int Wd,
                                float scale)
{
  long idx = (long)blockIdx.x * blockDim.x + threadIdx.x;
  long total = (long)B * C * Hd * Wd;
  if (idx >= total) return;
  long p = idx;
  int x = (int)(p % Wd); p /= Wd;
  int y = (int)(p % Hd); p /= Hd;
  int c = (int)(p % C);  int b = (int)(p / C);
  float ys = (Hd > 1) ? (float)y * (float)(Hs - 1) / (float)(Hd - 1) : 0.f;
  float xs = (Wd > 1) ? (float)x * (float)(Ws - 1) / (float)(Wd - 1) : 0.f;
  int y0 = (int)floorf(ys); int y1 = min(y0 + 1, Hs - 1);
  int x0 = (int)floorf(xs); int x1 = min(x0 + 1, Ws - 1);
  float wy = ys - (float)y0, wx = xs - (float)x0;
  const float* s = src + ((long)b * C + c) * (long)Hs * Ws;
  float v = s[(long)y0 * Ws + x0] * (1.f - wy) * (1.f - wx)
          + s[(long)y1 * Ws + x0] * wy * (1.f - wx)
          + s[(long)y0 * Ws + x1] * (1.f - wy) * wx
          + s[(long)y1 * Ws + x1] * wy * wx;
  dst[idx] = v * scale;
}

__global__ void copy_ch_kernel(const float* __restrict__ src,
                               float* __restrict__ dst,
                               int B, int Cc, int H, int W,
                               int sCt, int soff, int dCt, int doff)
{
  long idx = (long)blockIdx.x * blockDim.x + threadIdx.x;
  long total = (long)B * Cc * H * W;
  if (idx >= total) return;
  long p = idx;
  int x = (int)(p % W); p /= W;
  int y = (int)(p % H); p /= H;
  int c = (int)(p % Cc); int b = (int)(p / Cc);
  dst[(((long)b * dCt + doff + c) * H + y) * W + x] =
      src[(((long)b * sCt + soff + c) * H + y) * W + x];
}

__global__ void scale_flow_kernel(float* __restrict__ f, int B, int H, int W,
                                  float s0, float s1)
{
  long idx = (long)blockIdx.x * blockDim.x + threadIdx.x;
  long total = (long)B * 2 * H * W;
  if (idx >= total) return;
  int c = (int)((idx / ((long)H * W)) % 2);
  f[idx] *= (c == 0) ? s0 : s1;
}

__global__ void add_inplace_kernel(float* __restrict__ a,
                                   const float* __restrict__ b, long n)
{
  long idx = (long)blockIdx.x * blockDim.x + threadIdx.x;
  if (idx < n) a[idx] += b[idx];
}

__global__ void zero_kernel(float* __restrict__ a, long n)
{
  long idx = (long)blockIdx.x * blockDim.x + threadIdx.x;
  if (idx < n) a[idx] = 0.f;
}

// ---------------------------------------------------------------------------
// Host orchestration
// ---------------------------------------------------------------------------

static inline long cdivl(long a, long b) { return (a + b - 1) / b; }

extern "C" void kernel_launch(void* const* d_in, const int* in_sizes, int n_in,
                              void* d_out, int out_size, void* d_ws, size_t ws_size,
                              hipStream_t stream) {
  (void)in_sizes; (void)n_in; (void)out_size; (void)ws_size;
  const int B = 2, h_im = 384, w_im = 512;

  const float* events1 = (const float*)d_in[0];
  const float* events2 = (const float*)d_in[1];

  // params flattened in jax pytree (alphabetical) order:
  //   context(7), conv1x1(5), feat(6 blocks x 2), flow_est(6); 'b' before 'w'.
  const int base_ctx = 2, base_1x1 = 16, base_feat = 26, base_fest = 50;
  auto PB = [&](int base, int i) { return (const float*)d_in[base + 2 * i]; };
  auto PW = [&](int base, int i) { return (const float*)d_in[base + 2 * i + 1]; };

  // ---- workspace bump allocator (floats) ----
  float* ws = (float*)d_ws;
  size_t off = 0;
  auto alloc = [&](size_t n) { float* p = ws + off; off += n; return p; };

  const int CH[7] = {5, 16, 32, 64, 96, 128, 196};
  const int LHt[6] = {192, 96, 48, 24, 12, 6};
  const int LWt[6] = {256, 128, 64, 32, 16, 8};
  float* f1[6]; float* f2[6];
  for (int j = 0; j < 6; ++j) f1[j] = alloc((size_t)B * CH[j + 1] * LHt[j] * LWt[j]);
  for (int j = 0; j < 6; ++j) f2[j] = alloc((size_t)B * CH[j + 1] * LHt[j] * LWt[j]);
  float* flowA = alloc((size_t)B * 2 * 96 * 128);
  float* flowB = alloc((size_t)B * 2 * 96 * 128);
  float* delta = alloc((size_t)B * 2 * 96 * 128);
  float* x2w   = alloc((size_t)B * 32 * 96 * 128);          // max warped feature
  float* bufA  = alloc((size_t)B * 565 * 96 * 128);
  float* bufB  = alloc((size_t)B * 565 * 96 * 128);
  bf16_t* packW = (bf16_t*)alloc(524288);                   // 1,048,576 bf16

  auto egrid = [](long n) { return dim3((unsigned)cdivl(n, 256), 1, 1); };

  auto conv = [&](const float* in, int Bn, int Cin, int H, int W,
                  const float* w, const float* bias, int Cout, int k,
                  int stride, int dil, int relu,
                  float* out, int coff, int Ctot) {
    int K = Cin * k * k;
    int Kp = (K + 31) / 32 * 32;
    int Coutp = (Cout + 63) / 64 * 64;
    long nw = (long)Coutp * Kp;
    pack_weights_kernel<<<egrid(nw), 256, 0, stream>>>(w, packW, Cout, K, Coutp, Kp);
    int Hout = (stride == 2) ? H / 2 : H;
    int Wout = (stride == 2) ? W / 2 : W;
    long Npix = (long)Bn * Hout * Wout;
    dim3 grid((unsigned)cdivl(Npix, 32 * CONV_WAVES), (unsigned)(Coutp / 64), 1);
    int pad = (k - 1) * dil / 2;
    if (k == 1)
      conv2d_wmma_kernel<1><<<grid, dim3(32 * CONV_WAVES, 1, 1), 0, stream>>>(
          in, packW, bias, out, Bn, Cin, H, W, Cout, Hout, Wout,
          stride, dil, pad, relu, coff, Ctot, Kp);
    else
      conv2d_wmma_kernel<3><<<grid, dim3(32 * CONV_WAVES, 1, 1), 0, stream>>>(
          in, packW, bias, out, Bn, Cin, H, W, Cout, Hout, Wout,
          stride, dil, pad, relu, coff, Ctot, Kp);
  };

  // ---- feature pyramids ----
  auto pyramid = [&](const float* ev, float** fs) {
    const float* xin = ev; int cin = 5, H = 384, W = 512;
    for (int j = 0; j < 6; ++j) {
      int cout = CH[j + 1]; int Ho = H / 2, Wo = W / 2;
      conv(xin, B, cin, H, W, PW(base_feat, 2 * j), PB(base_feat, 2 * j),
           cout, 3, 2, 1, 1, bufB, 0, cout);
      conv(bufB, B, cout, Ho, Wo, PW(base_feat, 2 * j + 1), PB(base_feat, 2 * j + 1),
           cout, 3, 1, 1, 1, fs[j], 0, cout);
      xin = fs[j]; cin = cout; H = Ho; W = Wo;
    }
  };
  pyramid(events1, f1);
  pyramid(events2, f2);

  // ---- coarse-to-fine flow loop ----
  const int Cl[5] = {196, 128, 96, 64, 32};
  const int Hl[5] = {6, 12, 24, 48, 96};
  const int Wl[5] = {8, 16, 32, 64, 128};
  const int FO[5] = {128, 128, 96, 64, 32};            // flow_est outs (dense part)
  const int XO[6] = {128, 128, 128, 96, 64, 32};       // context outs
  const int XD[6] = {1, 2, 4, 8, 16, 1};               // context dilations

  float* flow = flowA; float* flowT = flowB;

  for (int l = 0; l < 5; ++l) {
    int C = Cl[l], H = Hl[l], W = Wl[l];
    const float* x1 = f1[5 - l];
    const float* x2 = f2[5 - l];
    const float* x2wp;
    long nf = (long)B * 2 * H * W;

    if (l == 0) {
      zero_kernel<<<egrid(nf), 256, 0, stream>>>(flow, nf);
      x2wp = x2;
    } else {
      upsample_kernel<<<egrid(nf), 256, 0, stream>>>(flow, flowT, B, 2,
                                                     Hl[l - 1], Wl[l - 1], H, W, 1.0f);
      float* t = flow; flow = flowT; flowT = t;
      long nw = (long)B * C * H * W;
      warp_kernel<<<egrid(nw), 256, 0, stream>>>(x2, flow, x2w, B, C, H, W, h_im, w_im);
      x2wp = x2w;
    }

    // x = concat([corr(81), conv1x1(x1)(32), flow_local(2)]) -> bufA, 115 ch
    long nc = (long)B * 81 * H * W;
    cost_volume_kernel<<<egrid(nc), 256, 0, stream>>>(x1, x2wp, bufA, B, C, H, W, 115);
    conv(x1, B, C, H, W, PW(base_1x1, l), PB(base_1x1, l),
         32, 1, 1, 1, 1, bufA, 81, 115);
    float usL = (float)W / ((float)w_im * DIV_FLOW);
    float vsL = (float)H / ((float)h_im * DIV_FLOW);
    scale_flow_kernel<<<egrid(nf), 256, 0, stream>>>(flow, B, H, W, usL, vsL);
    copy_ch_kernel<<<egrid(nf), 256, 0, stream>>>(flow, bufA, B, 2, H, W, 2, 0, 115, 113);

    // dense flow estimator: x = concat([conv(x), x])
    float* cur = bufA; float* oth = bufB; int curC = 115;
    for (int i = 0; i < 5; ++i) {
      int co = FO[i]; int nC = co + curC;
      conv(cur, B, curC, H, W, PW(base_fest, i), PB(base_fest, i),
           co, 3, 1, 1, 1, oth, 0, nC);
      long ncp = (long)B * curC * H * W;
      copy_ch_kernel<<<egrid(ncp), 256, 0, stream>>>(cur, oth, B, curC, H, W,
                                                     curC, 0, nC, co);
      float* t = cur; cur = oth; oth = t; curC = nC;
    }
    // flow += conv(x, flow_est[-1]) (no relu)
    conv(cur, B, curC, H, W, PW(base_fest, 5), PB(base_fest, 5),
         2, 3, 1, 1, 0, delta, 0, 2);
    add_inplace_kernel<<<egrid(nf), 256, 0, stream>>>(flow, delta, nf);

    // c = concat([x(563), flow(2)]) -> oth (565 ch)
    long nx = (long)B * curC * H * W;
    copy_ch_kernel<<<egrid(nx), 256, 0, stream>>>(cur, oth, B, curC, H, W,
                                                  curC, 0, 565, 0);
    copy_ch_kernel<<<egrid(nf), 256, 0, stream>>>(flow, oth, B, 2, H, W,
                                                  2, 0, 565, 563);
    float* src = oth; float* dst = cur; int sc = 565;
    for (int i = 0; i < 6; ++i) {
      conv(src, B, sc, H, W, PW(base_ctx, i), PB(base_ctx, i),
           XO[i], 3, 1, XD[i], 1, dst, 0, XO[i]);
      float* t = src; src = dst; dst = t; sc = XO[i];
    }
    conv(src, B, 32, H, W, PW(base_ctx, 6), PB(base_ctx, 6),
         2, 3, 1, 1, 0, delta, 0, 2);
    add_inplace_kernel<<<egrid(nf), 256, 0, stream>>>(flow, delta, nf);

    // rescale to global units, then upsampled prediction into d_out
    float usG = ((float)w_im * DIV_FLOW) / (float)W;
    float vsG = ((float)h_im * DIV_FLOW) / (float)H;
    scale_flow_kernel<<<egrid(nf), 256, 0, stream>>>(flow, B, H, W, usG, vsG);
    float* pred = (float*)d_out + (long)l * B * 2 * h_im * w_im;
    long np = (long)B * 2 * h_im * w_im;
    upsample_kernel<<<egrid(np), 256, 0, stream>>>(flow, pred, B, 2, H, W,
                                                   h_im, w_im, 1.0f / DIV_FLOW);
  }
}
